// GeoSlotV2_14972255994599
// MI455X (gfx1250) — matrix operand order repaired and checked by reading, not compiled, not run on hardware
//
#include <hip/hip_runtime.h>
#include <math.h>

// ---------------------------------------------------------------------------
// Types for CDNA5 WMMA
// ---------------------------------------------------------------------------
typedef __attribute__((ext_vector_type(16))) _Float16 v16h;
typedef __attribute__((ext_vector_type(8)))  _Float16 h8;
typedef __attribute__((ext_vector_type(4)))  _Float16 h4;
typedef __attribute__((ext_vector_type(8)))  float    v8f;

#define ACT_NONE 0
#define ACT_GELU 1
#define ACT_RELU 2

// Problem constants (from the reference)
#define M_ROWS   16384      // B*K = 2048*8
#define DSL      256        // D
#define NPIX     1024       // H*W = 32*32

__device__ __forceinline__ float gelu_exact(float x) {
    return 0.5f * x * (1.0f + erff(x * 0.70710678118654752f));
}

// ---------------------------------------------------------------------------
// WMMA GEMM: Y[M x N] = act(X[M x K] @ W[K x N] + bias) (+ resid)
//   X row stride = lda (f32), W row stride = Ndim (f32), Y row stride = ldo.
//   Block tile 128x128, 256 threads = 8 waves; wave tile 32x64 (2x4 WMMA tiles).
//   f32 -> f16 conversion during LDS staging; v_wmma_f32_16x16x32_f16 compute.
//   Double-buffered LDS; register-staged global loads; packed b64 LDS stores.
// ---------------------------------------------------------------------------
template<int ACT>
__launch_bounds__(256)
__global__ void gemm_wmma(const float* __restrict__ X, int lda,
                          const float* __restrict__ W, int Ndim,
                          const float* __restrict__ bias,
                          const float* __restrict__ resid,
                          float* __restrict__ Y, int ldo,
                          int Kdim)
{
    // padded stride 40 halves = 80 B: keeps every fragment chunk 16B aligned
    __shared__ _Float16 As[2][128 * 40];
    __shared__ _Float16 Bs[2][128 * 40];

    const int tid     = threadIdx.x;
    const int wave    = tid >> 5;
    const int lane    = tid & 31;
    const int half_id = lane >> 4;   // 0: lanes 0-15, 1: lanes 16-31
    const int lrow    = lane & 15;
    const int wm      = wave & 3;    // 4 row groups of 32
    const int wn      = wave >> 2;   // 2 col groups of 64
    const int mBase   = blockIdx.x * 128;
    const int nBase   = blockIdx.y * 128;

    // B staging: each thread owns a 4(k) x 4(n) sub-block of the 32x128 tile
    const int b_kg = (tid >> 5) << 2;   // k base: wave index * 4  (0..28)
    const int b_ng = (tid & 31) << 2;   // n base: lane * 4        (0..124)

    v8f acc[2][4];
    const v8f vzero = {0.f, 0.f, 0.f, 0.f, 0.f, 0.f, 0.f, 0.f};
#pragma unroll
    for (int a = 0; a < 2; ++a)
#pragma unroll
        for (int b = 0; b < 4; ++b) acc[a][b] = vzero;

    float4 av[4];
    float  br[4][4];   // br[kk][nn]

    // ---- tile loaders / storers -------------------------------------------
    auto load_tile = [&](int k0) {
        // A: 128 rows x 32 k as 128x8 float4s; thread t handles 4 of them
#pragma unroll
        for (int t = 0; t < 4; ++t) {
            int i   = tid + t * 256;
            int row = i >> 3;
            int c4  = (i & 7) << 2;
            av[t] = *(const float4*)(X + (size_t)(mBase + row) * lda + k0 + c4);
        }
        // B: 4 coalesced float4 rows of the thread's 4x4 sub-block
#pragma unroll
        for (int kk = 0; kk < 4; ++kk) {
            float4 v = *(const float4*)(W + (size_t)(k0 + b_kg + kk) * Ndim + nBase + b_ng);
            br[kk][0] = v.x; br[kk][1] = v.y; br[kk][2] = v.z; br[kk][3] = v.w;
        }
    };
    auto store_tile = [&](int buf) {
#pragma unroll
        for (int t = 0; t < 4; ++t) {
            int i   = tid + t * 256;
            int row = i >> 3;
            int c4  = (i & 7) << 2;
            h4 h;
            h[0] = (_Float16)av[t].x; h[1] = (_Float16)av[t].y;
            h[2] = (_Float16)av[t].z; h[3] = (_Float16)av[t].w;
            *(h4*)&As[buf][row * 40 + c4] = h;
        }
        // B transpose in registers -> 4 packed b64 stores (8B aligned)
#pragma unroll
        for (int nn = 0; nn < 4; ++nn) {
            h4 h;
            h[0] = (_Float16)br[0][nn]; h[1] = (_Float16)br[1][nn];
            h[2] = (_Float16)br[2][nn]; h[3] = (_Float16)br[3][nn];
            *(h4*)&Bs[buf][(b_ng + nn) * 40 + b_kg] = h;
        }
    };
    auto compute = [&](int buf) {
        v16h afrag[2], bfrag[4];
#pragma unroll
        for (int fm = 0; fm < 2; ++fm) {
            int row = wm * 32 + fm * 16 + lrow;
            // lanes<16: K=0..7 then 16..23 ; lanes>=16: K=8..15 then 24..31
            h8 lo = *(const h8*)&As[buf][row * 40 + half_id * 8];
            h8 hi = *(const h8*)&As[buf][row * 40 + 16 + half_id * 8];
            afrag[fm] = __builtin_shufflevector(lo, hi,
                0,1,2,3,4,5,6,7,8,9,10,11,12,13,14,15);
        }
#pragma unroll
        for (int fn = 0; fn < 4; ++fn) {
            int n = wn * 64 + fn * 16 + lrow;
            // lanes<16: K=0..7 then 16..23 ; lanes>=16: K=8..15 then 24..31
            h8 lo = *(const h8*)&Bs[buf][n * 40 + half_id * 8];
            h8 hi = *(const h8*)&Bs[buf][n * 40 + 16 + half_id * 8];
            bfrag[fn] = __builtin_shufflevector(lo, hi,
                0,1,2,3,4,5,6,7,8,9,10,11,12,13,14,15);
        }
#pragma unroll
        for (int fm = 0; fm < 2; ++fm)
#pragma unroll
            for (int fn = 0; fn < 4; ++fn)
                acc[fm][fn] = __builtin_amdgcn_wmma_f32_16x16x32_f16(
                    false, afrag[fm], false, bfrag[fn],
                    (short)0, acc[fm][fn], false, false);
    };

    // ---- software-pipelined main loop -------------------------------------
    const int nk = Kdim >> 5;
    load_tile(0);
    store_tile(0);
    for (int kt = 0; kt < nk; ++kt) {
        const int cur = kt & 1;
        if (kt + 1 < nk) load_tile((kt + 1) << 5);   // issue next tile's loads
        __syncthreads();                             // buf[cur] ready
        compute(cur);                                // WMMA from buf[cur]
        if (kt + 1 < nk) store_tile(cur ^ 1);        // stage next tile
    }

    // ---- epilogue: bias + activation + optional residual ------------------
#pragma unroll
    for (int fm = 0; fm < 2; ++fm)
#pragma unroll
        for (int fn = 0; fn < 4; ++fn) {
            int col = nBase + wn * 64 + fn * 16 + lrow;   // logical GEMM column
#pragma unroll
            for (int r = 0; r < 8; ++r) {
                int row = mBase + wm * 32 + fm * 16 + half_id * 8 + r;
                float v = acc[fm][fn][r];
                if (bias) v += bias[col];
                if (ACT == ACT_GELU) v = gelu_exact(v);
                if (ACT == ACT_RELU) v = fmaxf(v, 0.0f);
                if (resid) v += resid[(size_t)row * DSL + col];
                Y[(size_t)row * ldo + col] = v;
            }
        }
}

// ---------------------------------------------------------------------------
// Centroids: one block per (b,k) row of attn (N=1024). cent[row] = (cx,cy)
// ---------------------------------------------------------------------------
__global__ void centroid_kernel(const float* __restrict__ attn,
                                float* __restrict__ cent)
{
    __shared__ float sA[256], sX[256], sY[256];
    const int row = blockIdx.x;
    const int tid = threadIdx.x;
    float sa = 0.f, sx = 0.f, sy = 0.f;
#pragma unroll
    for (int t = 0; t < 4; ++t) {
        int n = tid + t * 256;
        float a = attn[(size_t)row * NPIX + n];
        int xi = n & 31;          // col -> x coordinate
        int yi = n >> 5;          // row -> y coordinate
        sa += a;
        sx += a * ((float)xi * (1.0f / 31.0f));
        sy += a * ((float)yi * (1.0f / 31.0f));
    }
    sA[tid] = sa; sX[tid] = sx; sY[tid] = sy;
    __syncthreads();
    for (int off = 128; off > 0; off >>= 1) {
        if (tid < off) {
            sA[tid] += sA[tid + off];
            sX[tid] += sX[tid + off];
            sY[tid] += sY[tid + off];
        }
        __syncthreads();
    }
    if (tid == 0) {
        float inv = 1.0f / (sA[0] + 1e-8f);
        cent[(size_t)row * 2 + 0] = sX[0] * inv;
        cent[(size_t)row * 2 + 1] = sY[0] * inv;
    }
}

// ---------------------------------------------------------------------------
// Sinusoidal encoding: enc[row][c], c = f*64 + (sin? j : 32+j), feats=(cx,cy,.1,.1)
// ---------------------------------------------------------------------------
__global__ void encode_kernel(const float* __restrict__ cent,
                              float* __restrict__ enc)
{
    const int row = blockIdx.x;
    const int c   = threadIdx.x;            // 0..255
    const int f   = c >> 6;                 // feature 0..3
    const int t   = c & 63;
    const int j   = t & 31;                 // frequency index
    float feat = (f == 0) ? cent[(size_t)row * 2 + 0]
               : (f == 1) ? cent[(size_t)row * 2 + 1] : 0.1f;
    // freqs = 10000^(-(2j)/64)
    float freq = __expf(-(float)(2 * j) * (9.210340371976184f / 64.0f));
    float ang  = feat * freq * 3.14159f;
    enc[(size_t)row * DSL + c] = (t < 32) ? sinf(ang) : cosf(ang);
}

// ---------------------------------------------------------------------------
// LayerNorm over D=256: one wave per row (wave32, shfl_xor reductions)
// ---------------------------------------------------------------------------
__global__ void ln_kernel(const float* __restrict__ X,
                          const float* __restrict__ g,
                          const float* __restrict__ b,
                          float* __restrict__ Y)
{
    const int wave = threadIdx.x >> 5;
    const int lane = threadIdx.x & 31;
    const int row  = blockIdx.x * 8 + wave;
    const float* x = X + (size_t)row * DSL;
    float v[8];
    float s = 0.f;
#pragma unroll
    for (int r = 0; r < 8; ++r) { v[r] = x[lane + 32 * r]; s += v[r]; }
#pragma unroll
    for (int off = 16; off > 0; off >>= 1) s += __shfl_xor(s, off, 32);
    float mean = s * (1.0f / 256.0f);
    float var = 0.f;
#pragma unroll
    for (int r = 0; r < 8; ++r) { float d = v[r] - mean; var += d * d; }
#pragma unroll
    for (int off = 16; off > 0; off >>= 1) var += __shfl_xor(var, off, 32);
    var *= (1.0f / 256.0f);
    float rstd = rsqrtf(var + 1e-5f);
    float* y = Y + (size_t)row * DSL;
#pragma unroll
    for (int r = 0; r < 8; ++r) {
        int c = lane + 32 * r;
        y[c] = (v[r] - mean) * rstd * g[c] + b[c];
    }
}

// ---------------------------------------------------------------------------
// SSM gate: H[:, :512] = silu(z) * sigmoid(gate), in place (row stride 1024)
// ---------------------------------------------------------------------------
__global__ void ssm_act_kernel(float* __restrict__ Hm)
{
    size_t idx = (size_t)blockIdx.x * 256 + threadIdx.x;  // over 16384*512
    size_t row = idx >> 9;
    int    j   = (int)(idx & 511);
    float z = Hm[row * 1024 + j];
    float g = Hm[row * 1024 + 512 + j];
    float sz = 1.0f / (1.0f + __expf(-z));
    float sg = 1.0f / (1.0f + __expf(-g));
    Hm[row * 1024 + j] = z * sz * sg;
}

// ---------------------------------------------------------------------------
// FGW matching + log-domain Sinkhorn: one 64-thread block per batch
// ---------------------------------------------------------------------------
__global__ void fgw_kernel(const float* __restrict__ pq,
                           const float* __restrict__ pr,
                           const float* __restrict__ cq,
                           const float* __restrict__ cr,
                           const float* __restrict__ log_eps,
                           float* __restrict__ outT,
                           float* __restrict__ outCost,
                           float* __restrict__ outSim)
{
    __shared__ float qs[8 * 256], rs[8 * 256];
    __shared__ float Cm[64], Sq[64], Sr[64], Sq2[64], Sr2[64];
    __shared__ float Tm[64], Cf[64], lK[64], U[64];
    __shared__ float lu[8], lv[8], rsum[8], csum[8];
    __shared__ float red[64];

    const int bat = blockIdx.x;
    const int tid = threadIdx.x;
    const int i = tid >> 3, j = tid & 7;

    for (int t = tid; t < 2048; t += 64) {
        qs[t] = pq[(size_t)bat * 2048 + t];
        rs[t] = pr[(size_t)bat * 2048 + t];
    }
    __syncthreads();

    // feature cost C[i][j]
    float accd = 0.f;
    for (int d = 0; d < 256; ++d) {
        float df = qs[i * 256 + d] - rs[j * 256 + d];
        accd += df * df;
    }
    Cm[tid] = sqrtf(fmaxf(accd, 1e-6f));

    // structure matrices
    {
        float dx = cq[(size_t)bat * 16 + i * 2 + 0] - cq[(size_t)bat * 16 + j * 2 + 0];
        float dy = cq[(size_t)bat * 16 + i * 2 + 1] - cq[(size_t)bat * 16 + j * 2 + 1];
        float s = sqrtf(fmaxf(dx * dx + dy * dy, 1e-6f));
        Sq[tid] = s; Sq2[tid] = s * s;
        dx = cr[(size_t)bat * 16 + i * 2 + 0] - cr[(size_t)bat * 16 + j * 2 + 0];
        dy = cr[(size_t)bat * 16 + i * 2 + 1] - cr[(size_t)bat * 16 + j * 2 + 1];
        s = sqrtf(fmaxf(dx * dx + dy * dy, 1e-6f));
        Sr[tid] = s; Sr2[tid] = s * s;
    }
    Tm[tid] = 1.0f / 64.0f;

    float eps = fminf(fmaxf(__expf(log_eps[0]), 0.01f), 0.5f);
    float rho = 0.1f / (0.1f + eps);
    const float log_mu = logf(0.125f);
    __syncthreads();

    for (int it = 0; it < 5; ++it) {
        if (tid < 8) {
            float s = 0.f, c = 0.f;
            for (int m = 0; m < 8; ++m) { s += Tm[tid * 8 + m]; c += Tm[m * 8 + tid]; }
            rsum[tid] = s; csum[tid] = c;
        }
        // U = T @ Sr (uses Tm from previous barrier-synced write)
        float u = 0.f;
        for (int m = 0; m < 8; ++m) u += Tm[i * 8 + m] * Sr[m * 8 + j];
        U[tid] = u;
        __syncthreads();

        float t1 = 0.f, t2 = 0.f, t3 = 0.f;
        for (int l = 0; l < 8; ++l) {
            t1 += Sq2[i * 8 + l] * rsum[l];
            t3 += Sq[i * 8 + l] * U[l * 8 + j];
        }
        for (int m = 0; m < 8; ++m) t2 += csum[m] * Sr2[m * 8 + j];
        float cf = 0.5f * Cm[tid] + 0.5f * (t1 + t2 - 2.0f * t3);  // LAM = 0.5
        Cf[tid] = cf;
        lK[tid] = -cf / eps;
        if (tid < 8) { lu[tid] = 0.f; lv[tid] = 0.f; }
        __syncthreads();

        for (int s = 0; s < 10; ++s) {
            if (tid < 8) {
                float mx = -1e30f;
                for (int jj = 0; jj < 8; ++jj) mx = fmaxf(mx, lK[tid * 8 + jj] + lv[jj]);
                float se = 0.f;
                for (int jj = 0; jj < 8; ++jj) se += __expf(lK[tid * 8 + jj] + lv[jj] - mx);
                lu[tid] = rho * (log_mu - (mx + __logf(se)));
            }
            __syncthreads();
            if (tid < 8) {
                float mx = -1e30f;
                for (int ii = 0; ii < 8; ++ii) mx = fmaxf(mx, lK[ii * 8 + tid] + lu[ii]);
                float se = 0.f;
                for (int ii = 0; ii < 8; ++ii) se += __expf(lK[ii * 8 + tid] + lu[ii] - mx);
                lv[tid] = rho * (log_mu - (mx + __logf(se)));
            }
            __syncthreads();
        }
        Tm[tid] = __expf(lu[i] + lK[tid] + lv[j]);
        __syncthreads();
    }

    red[tid] = Tm[tid] * Cf[tid];
    __syncthreads();
    for (int off = 32; off > 0; off >>= 1) {
        if (tid < off) red[tid] += red[tid + off];
        __syncthreads();
    }
    outT[(size_t)bat * 64 + tid] = Tm[tid];
    if (tid == 0) {
        float c = red[0];
        outCost[bat] = c;
        outSim[bat] = 1.0f / (1.0f + __expf(c));   // sigmoid(-cost)
    }
}

// ---------------------------------------------------------------------------
// Host launcher
// ---------------------------------------------------------------------------
extern "C" void kernel_launch(void* const* d_in, const int* in_sizes, int n_in,
                              void* d_out, int out_size, void* d_ws, size_t ws_size,
                              hipStream_t stream)
{
    (void)in_sizes; (void)n_in; (void)out_size; (void)ws_size;

    const float* slots_q = (const float*)d_in[0];
    const float* slots_r = (const float*)d_in[1];
    const float* attn    = (const float*)d_in[2];
    const float* cent_r  = (const float*)d_in[3];
    const float* pm_w1   = (const float*)d_in[4];
    const float* pm_b1   = (const float*)d_in[5];
    const float* pm_w2   = (const float*)d_in[6];
    const float* pm_b2   = (const float*)d_in[7];
    const float* ssm_ng  = (const float*)d_in[8];
    const float* ssm_nb  = (const float*)d_in[9];
    const float* ssm_wi  = (const float*)d_in[10];
    const float* ssm_bi  = (const float*)d_in[11];
    const float* ssm_wo  = (const float*)d_in[12];
    const float* ssm_bo  = (const float*)d_in[13];
    const float* merge_w = (const float*)d_in[14];
    const float* merge_b = (const float*)d_in[15];
    const float* norm_g  = (const float*)d_in[16];
    const float* norm_b  = (const float*)d_in[17];
    const float* ffn_lg  = (const float*)d_in[18];
    const float* ffn_lb  = (const float*)d_in[19];
    const float* ffn_w1  = (const float*)d_in[20];
    const float* ffn_b1  = (const float*)d_in[21];
    const float* ffn_w2  = (const float*)d_in[22];
    const float* ffn_b2  = (const float*)d_in[23];
    const float* log_eps = (const float*)d_in[24];
    const float* cp_w1   = (const float*)d_in[25];
    const float* cp_b1   = (const float*)d_in[26];
    const float* cp_w2   = (const float*)d_in[27];
    const float* cp_b2   = (const float*)d_in[28];

    float* out      = (float*)d_out;
    float* out_slot = out;                       // 16384*256
    float* out_cent = out_slot + 4194304;        // 16384*2
    float* out_sim  = out_cent + 32768;          // 2048
    float* out_T    = out_sim  + 2048;           // 2048*64
    float* out_cost = out_T    + 131072;         // 2048

    float* ws   = (float*)d_ws;
    float* ws0  = ws;                 // 16384*256  : current slots
    float* ws1  = ws0  + 4194304;     // 16384*256  : LN / hidden temp
    float* wsH  = ws1  + 4194304;     // 16384*1024 : wide hidden
    float* wsFB = wsH  + 16777216;    // 16384*512  : concat(f, bk) / pr
    float* wsT  = wsFB + 8388608;     // 16384*256  : enc / pre-LN / pq

    const dim3 blk(256);
    const dim3 g2(128, 2);   // N = 256
    const dim3 g8(128, 8);   // N = 1024

    // ---- SlotSpatialEncoder ----
    centroid_kernel<<<16384, 256, 0, stream>>>(attn, out_cent);
    encode_kernel<<<16384, 256, 0, stream>>>(out_cent, wsT);
    gemm_wmma<ACT_GELU><<<g2, blk, 0, stream>>>(wsT, 256, pm_w1, 256, pm_b1,
                                                nullptr, ws1, 256, 256);
    gemm_wmma<ACT_NONE><<<g2, blk, 0, stream>>>(ws1, 256, pm_w2, 256, pm_b2,
                                                slots_q, ws0, 256, 256);

    // ---- GraphMambaLayer x2 (Hilbert permutation cancels: SSM is pointwise) ----
    for (int i = 0; i < 2; ++i) {
        for (int d = 0; d < 2; ++d) {
            const int id = i * 2 + d;
            ln_kernel<<<2048, 256, 0, stream>>>(ws0, ssm_ng + id * 256,
                                                ssm_nb + id * 256, ws1);
            gemm_wmma<ACT_NONE><<<g8, blk, 0, stream>>>(
                ws1, 256, ssm_wi + (size_t)id * 256 * 1024, 1024,
                ssm_bi + id * 1024, nullptr, wsH, 1024, 256);
            ssm_act_kernel<<<32768, 256, 0, stream>>>(wsH);
            gemm_wmma<ACT_NONE><<<g2, blk, 0, stream>>>(
                wsH, 1024, ssm_wo + (size_t)id * 512 * 256, 256,
                ssm_bo + id * 256, ws0, wsFB + d * 256, 512, 512);
        }
        gemm_wmma<ACT_NONE><<<g2, blk, 0, stream>>>(
            wsFB, 512, merge_w + (size_t)i * 512 * 256, 256,
            merge_b + i * 256, ws0, wsT, 256, 512);
        ln_kernel<<<2048, 256, 0, stream>>>(wsT, norm_g + i * 256,
                                            norm_b + i * 256, ws0);
        ln_kernel<<<2048, 256, 0, stream>>>(ws0, ffn_lg + i * 256,
                                            ffn_lb + i * 256, ws1);
        gemm_wmma<ACT_GELU><<<g8, blk, 0, stream>>>(
            ws1, 256, ffn_w1 + (size_t)i * 256 * 1024, 1024,
            ffn_b1 + i * 1024, nullptr, wsH, 1024, 256);
        float* dst = (i == 1) ? out_slot : ws0;
        gemm_wmma<ACT_NONE><<<g2, blk, 0, stream>>>(
            wsH, 1024, ffn_w2 + (size_t)i * 1024 * 256, 256,
            ffn_b2 + i * 256, ws0, dst, 256, 1024);
    }

    // ---- cost projections: pq from final slots (d_out), pr from slots_r ----
    gemm_wmma<ACT_RELU><<<g2, blk, 0, stream>>>(out_slot, 256, cp_w1, 256, cp_b1,
                                                nullptr, ws1, 256, 256);
    gemm_wmma<ACT_NONE><<<g2, blk, 0, stream>>>(ws1, 256, cp_w2, 256, cp_b2,
                                                nullptr, wsT, 256, 256);   // pq
    gemm_wmma<ACT_RELU><<<g2, blk, 0, stream>>>(slots_r, 256, cp_w1, 256, cp_b1,
                                                nullptr, ws1, 256, 256);
    gemm_wmma<ACT_NONE><<<g2, blk, 0, stream>>>(ws1, 256, cp_w2, 256, cp_b2,
                                                nullptr, wsFB, 256, 256);  // pr

    // ---- Fused Gromov-Wasserstein + Sinkhorn ----
    fgw_kernel<<<2048, 64, 0, stream>>>(wsT, wsFB, out_cent, cent_r, log_eps,
                                        out_T, out_cost, out_sim);
}
